// OrbitalCofactorAntiequivarianceLayer_14851996910245
// MI455X (gfx1250) — matrix operand order repaired and checked by reading, not compiled, not run on hardware
//
#include <hip/hip_runtime.h>

typedef __attribute__((ext_vector_type(16))) _Float16 v16h;
typedef __attribute__((ext_vector_type(8)))  float    v8f;

#define B_ 8192
#define S_ 2
#define N_ 8
#define D_ 256
#define I_ 4

#define WAVES_PER_BLOCK 8
#define BLOCK (WAVES_PER_BLOCK * 32)

// ---------------------------------------------------------------------------
// Prep kernel: build the per-lane WMMA B-fragments for W (zero-padded to 16
// orbital columns, converted to f16) once.  Layout in d_ws:
//   frag[(s*8 + kk)*32 + lane][t] , t = 0..15   (v16h per lane)
// B-fragment order chosen here for v_wmma_f32_16x16x32_f16 (contiguous k, same as A):
//   lane: col n = lane&15, half hi = lane>>4 ; element t: K = kk*32 + hi*16 + t
// ---------------------------------------------------------------------------
__global__ __launch_bounds__(512)
void build_bfrag_kernel(const float* __restrict__ W, _Float16* __restrict__ frag)
{
    const int tid  = threadIdx.x;         // 0..511
    const int s    = tid >> 8;
    const int kk   = (tid >> 5) & 7;
    const int lane = tid & 31;
    const int n    = lane & 15;
    const int hi   = lane >> 4;

    _Float16* dst = frag + (size_t)((s * 8 + kk) * 32 + lane) * 16;
#pragma unroll
    for (int t = 0; t < 16; ++t) {
        int   d  = kk * 32 + hi * 16 + t;
        float wv = (n < N_) ? W[(s * D_ + d) * N_ + n] : 0.0f;
        dst[t] = (_Float16)wv;
    }
}

// ---------------------------------------------------------------------------
// Main fused kernel: one wave owns 2 walkers (a 16-row WMMA M-tile).
// ---------------------------------------------------------------------------
__global__ __launch_bounds__(BLOCK)
void orbital_cofactor_kernel(const float* __restrict__ eq,
                             const float* __restrict__ r_ei,
                             const float* __restrict__ bias,
                             const float* __restrict__ decay,
                             const float* __restrict__ pi,
                             const v16h*  __restrict__ bfrags,
                             float* __restrict__ out)
{
    __shared__ float lds_mat[WAVES_PER_BLOCK][2][N_][N_];    // orbital matrices
    __shared__ float lds_anti[WAVES_PER_BLOCK][2][N_];
    __shared__ float lds_dist[WAVES_PER_BLOCK][16][I_];

    const int tid  = threadIdx.x;
    const int lane = tid & 31;
    const int wave = tid >> 5;
    const int hi   = lane >> 4;     // half-wave selector
    const int lj   = lane & 15;     // row (A) / column (B,C) index

    const int wavet = blockIdx.x * WAVES_PER_BLOCK + wave;
    const int w0    = wavet * 2;    // first of 2 walkers owned by this wave

    for (int s = 0; s < S_; ++s) {
        // ---- electron-ion distances: 16 rows x 4 ions per wave ----
        {
            int row = lane >> 1;                 // 0..15 (walker*8 + electron)
            int b   = w0 + (row >> 3);
            int m   = row & 7;
            const float* rp = r_ei + (size_t)((b * 16 + s * 8 + m) * I_) * 3;
#pragma unroll
            for (int t = 0; t < 2; ++t) {
                int i = (lane & 1) * 2 + t;
                float x = rp[i * 3 + 0], y = rp[i * 3 + 1], z = rp[i * 3 + 2];
                lds_dist[wave][row][i] = sqrtf(x * x + y * y + z * z);
            }
        }
        __syncthreads();

        // ---- lin = h @ W_s via v_wmma_f32_16x16x32_f16, K = 256 ----
        const int    mrow = lj;                           // A-matrix row M
        const int    bA   = w0 + (mrow >> 3);
        const float* hrow = eq + (size_t)(bA * 16 + s * 8 + (mrow & 7)) * D_;

        v8f acc = {};
#pragma unroll
        for (int kk = 0; kk < 8; ++kk) {
            const int kbase = kk * 32;

            // A fragment: K = {kbase+hi*16 .. +15}
            float ta[16];
            *(float4*)&ta[0]  = *(const float4*)(hrow + kbase + hi * 16);
            *(float4*)&ta[4]  = *(const float4*)(hrow + kbase + hi * 16 + 4);
            *(float4*)&ta[8]  = *(const float4*)(hrow + kbase + hi * 16 + 8);
            *(float4*)&ta[12] = *(const float4*)(hrow + kbase + hi * 16 + 12);
            v16h afrag;
#pragma unroll
            for (int j = 0; j < 16; ++j) afrag[j] = (_Float16)ta[j];

            // B fragment: one coalesced 32-byte vector load, no divergence
            v16h bfrag = bfrags[(s * 8 + kk) * 32 + lane];

            acc = __builtin_amdgcn_wmma_f32_16x16x32_f16(
                false, afrag, false, bfrag, (short)0, acc, false, false);
        }

        // ---- bias + envelope, scatter orbital matrices to LDS ----
        if (lj < N_) {
            const float bk = bias[s * N_ + lj];
            float pl[I_], dl[I_];
#pragma unroll
            for (int i = 0; i < I_; ++i) {
                pl[i] = pi[(s * I_ + i) * N_ + lj];
                dl[i] = decay[(s * I_ + i) * N_ + lj];
            }
#pragma unroll
            for (int v = 0; v < 8; ++v) {
                // C layout: lane lj holds (M = v + 8*hi, N = lj); walker = hi
                float env = 0.0f;
#pragma unroll
                for (int i = 0; i < I_; ++i)
                    env += pl[i] * __expf(-lds_dist[wave][hi * 8 + v][i] * dl[i]);
                lds_mat[wave][hi][v][lj] = (acc[v] + bk) * env;
            }
        }
        __syncthreads();

        // ---- anti[i] = M[i,0] * (-1)^i * det(minor(i,0)), one lane each ----
        if (lane < 16) {
            const int wk   = lane >> 3;
            const int iexc = lane & 7;
            float sub[7][7];
#pragma unroll
            for (int r = 0; r < 7; ++r) {
                int rr = (r < iexc) ? r : r + 1;   // delete row iexc
#pragma unroll
                for (int c = 0; c < 7; ++c)
                    sub[r][c] = lds_mat[wave][wk][rr][c + 1];  // delete col 0
            }
            float det = 1.0f;
#pragma unroll
            for (int p = 0; p < 7; ++p) {
                float piv = sub[p][p];
                det *= piv;
                float inv = 1.0f / piv;
#pragma unroll
                for (int r2 = p + 1; r2 < 7; ++r2) {
                    float f = sub[r2][p] * inv;
#pragma unroll
                    for (int c = p + 1; c < 7; ++c)
                        sub[r2][c] -= f * sub[p][c];
                }
            }
            float sgn = (iexc & 1) ? -1.0f : 1.0f;
            lds_anti[wave][wk][iexc] = lds_mat[wave][wk][iexc][0] * sgn * det;
        }
        __syncthreads();

        // ---- out[b,s,n,:] = h[b,s,n,:] * anti  (coalesced float4) ----
#pragma unroll
        for (int wk = 0; wk < 2; ++wk) {
            const int b = w0 + wk;
#pragma unroll
            for (int n = 0; n < N_; ++n) {
                const float  a  = lds_anti[wave][wk][n];
                const float* hp = eq  + (size_t)(b * 16 + s * 8 + n) * D_;
                float*       op = out + (size_t)s * B_ * N_ * D_
                                      + (size_t)(b * 8 + n) * D_;
#pragma unroll
                for (int c = 0; c < 2; ++c) {
                    int d = c * 128 + lane * 4;
                    float4 hv = *(const float4*)(hp + d);
                    float4 ov = make_float4(hv.x * a, hv.y * a, hv.z * a, hv.w * a);
                    *(float4*)(op + d) = ov;
                }
            }
        }
        __syncthreads();   // LDS reused by next spin iteration
    }
}

extern "C" void kernel_launch(void* const* d_in, const int* in_sizes, int n_in,
                              void* d_out, int out_size, void* d_ws, size_t ws_size,
                              hipStream_t stream) {
    const float* eq    = (const float*)d_in[0];  // (B, S*N, D)
    const float* r_ei  = (const float*)d_in[1];  // (B, S*N, I, 3)
    const float* W     = (const float*)d_in[2];  // (S, D, N)
    const float* bias  = (const float*)d_in[3];  // (S, N)
    const float* decay = (const float*)d_in[4];  // (S, I, N)
    const float* pi    = (const float*)d_in[5];  // (S, I, N)
    float*       out   = (float*)d_out;          // spin0 block then spin1 block

    _Float16* wsW = (_Float16*)d_ws;             // 16 KB of B-fragments

    hipLaunchKernelGGL(build_bfrag_kernel, dim3(1), dim3(512), 0, stream, W, wsW);

    dim3 grid(B_ / (2 * WAVES_PER_BLOCK));       // 512 blocks
    dim3 block(BLOCK);                           // 256 threads = 8 waves
    hipLaunchKernelGGL(orbital_cofactor_kernel, grid, block, 0, stream,
                       eq, r_ei, bias, decay, pi, (const v16h*)wsW, out);
}
